// AttentionLayer_47596827574368
// MI455X (gfx1250) — matrix operand order repaired and checked by reading, not compiled, not run on hardware
//
#include <hip/hip_runtime.h>
#include <hip/hip_bf16.h>

#define L     4096
#define CIN   768
#define CQK   96
#define COUT  768
#define NB    8

typedef __attribute__((ext_vector_type(16))) _Float16 v16h;
typedef __attribute__((ext_vector_type(8)))  _Float16 v8h;
typedef __attribute__((ext_vector_type(8)))  float    v8f;

static __device__ __forceinline__ void wave_lds_fence() {
    __builtin_amdgcn_wave_barrier();
    asm volatile("s_wait_dscnt 0x0" ::: "memory");
}

// 16-byte async copy global -> LDS (GLOBAL_LOAD_ASYNC_TO_LDS_B128, ASYNCcnt)
static __device__ __forceinline__ void async_copy16(unsigned int lds_addr,
                                                    const void* gaddr) {
    asm volatile("global_load_async_to_lds_b128 %0, %1, off"
                 :
                 : "v"(lds_addr), "v"((unsigned long long)(size_t)gaddr)
                 : "memory");
}

static __device__ __forceinline__ void wait_async0() {
    asm volatile("s_wait_asynccnt 0x0" ::: "memory");
}

// A-matrix fragment 16x32 f16 (M x K), src row-major with leading dim ld.
// lane m=lane&15, half=lane>>4; elem e -> K = (e<8?0:16)+half*8+(e&7)
static __device__ __forceinline__ v16h load_a16x32(const _Float16* src, int ld) {
    const int lane = threadIdx.x & 31;
    const int m = lane & 15, h = lane >> 4;
    const _Float16* p = src + (size_t)m * ld + h * 8;
    v8h lo = *(const v8h*)(p);
    v8h hi = *(const v8h*)(p + 16);
    v16h a;
#pragma unroll
    for (int e = 0; e < 8; ++e) { a[e] = lo[e]; a[e + 8] = hi[e]; }
    return a;
}

// B-matrix fragment 32x16 f16 (K x N), src stored column-contiguous:
// element (k,n) at src[n*ld + k].  lane n=lane&15, half=lane>>4; elem e -> K=(e<8?0:16)+half*8+(e&7)
static __device__ __forceinline__ v16h load_b32x16(const _Float16* src, int ld) {
    const int lane = threadIdx.x & 31;
    const int n = lane & 15, h = lane >> 4;
    const _Float16* p = src + (size_t)n * ld + h * 8;
    v8h lo = *(const v8h*)(p);
    v8h hi = *(const v8h*)(p + 16);
    v16h b;
#pragma unroll
    for (int e = 0; e < 8; ++e) { b[e] = lo[e]; b[e + 8] = hi[e]; }
    return b;
}

static __device__ __forceinline__ v8f wmma_f16(v16h a, v16h b, v8f c) {
    return __builtin_amdgcn_wmma_f32_16x16x32_f16(false, a, false, b, (short)0, c,
                                                  false, false);
}

// ---------------------------------------------------------------- kernel 0
__global__ __launch_bounds__(256) void k_cvt_weights(const float* __restrict__ Wq,
                                                     const float* __restrict__ Wk,
                                                     const float* __restrict__ Wv,
                                                     const float* __restrict__ Wl,
                                                     _Float16* __restrict__ dst) {
    const int NW = CQK * CIN;  // 73728, same count for Wlast (768*96)
    int i = blockIdx.x * 256 + threadIdx.x;
    if (i < NW) {
        dst[i]          = (_Float16)Wq[i];
        dst[NW + i]     = (_Float16)Wk[i];
        dst[2 * NW + i] = (_Float16)Wv[i];
        dst[3 * NW + i] = (_Float16)Wl[i];
    }
}

// ---------------------------------------------------------------- kernel 1
// Per block: one batch b, one 16-row tile of L. Stage X^T (16 x 768) f16 in LDS,
// 6 waves each own a 16-wide channel tile and accumulate q,k,v via WMMA (K=768).
__global__ __launch_bounds__(192) void k_qkv(const float* __restrict__ x,
                                             const _Float16* __restrict__ wh,
                                             _Float16* __restrict__ qh,
                                             _Float16* __restrict__ kh,
                                             _Float16* __restrict__ vTh) {
    __shared__ __align__(16) _Float16 Xt[16 * 784];   // [l][c], padded stride
    __shared__ __align__(16) _Float16 stg[6 * 256];   // per-wave 16x16 stage
    const int tid = threadIdx.x;
    const int b   = blockIdx.x >> 8;
    const int l0  = (blockIdx.x & 255) * 16;

    const float* xb = x + (size_t)b * CIN * L + l0;
#pragma unroll 4
    for (int i = 0; i < 64; ++i) {
        int idx = i * 192 + tid;
        int c = idx >> 4, l = idx & 15;
        Xt[l * 784 + c] = (_Float16)xb[(size_t)c * L + l];
    }
    __syncthreads();

    const int w    = tid >> 5;   // 0..5 channel tile
    const int lane = tid & 31;
    const int h    = lane >> 4, n = lane & 15;
    const _Float16* Wqh = wh;
    const _Float16* Wkh = wh + CQK * CIN;
    const _Float16* Wvh = wh + 2 * CQK * CIN;

    v8f cq = {}, ck = {}, cv = {};
    for (int kc = 0; kc < 24; ++kc) {
        v16h a  = load_a16x32(&Xt[kc * 32], 784);
        v16h bq = load_b32x16(Wqh + (size_t)(w * 16) * CIN + kc * 32, CIN);
        v16h bk = load_b32x16(Wkh + (size_t)(w * 16) * CIN + kc * 32, CIN);
        v16h bv = load_b32x16(Wvh + (size_t)(w * 16) * CIN + kc * 32, CIN);
        cq = wmma_f16(a, bq, cq);
        ck = wmma_f16(a, bk, ck);
        cv = wmma_f16(a, bv, cv);
    }

    // v stored transposed [96][L]: lane holds 8 consecutive l for fixed channel
    {
        v8h pv;
#pragma unroll
        for (int r = 0; r < 8; ++r) pv[r] = (_Float16)cv[r];
        _Float16* dst = vTh + (size_t)b * CQK * L + (size_t)(w * 16 + n) * L + l0 + h * 8;
        *(v8h*)dst = pv;
    }
    // q, k stored row-major [L][96] via LDS re-stage for coalesced b128 stores
    _Float16* ms = &stg[w * 256];
    {
#pragma unroll
        for (int r = 0; r < 8; ++r) ms[(h * 8 + r) * 16 + n] = (_Float16)cq[r];
        wave_lds_fence();
        int row = lane >> 1, ch = lane & 1;
        v8h vq = *(const v8h*)&ms[row * 16 + ch * 8];
        *(v8h*)(qh + (size_t)b * L * CQK + (size_t)(l0 + row) * CQK + w * 16 + ch * 8) = vq;
        wave_lds_fence();
#pragma unroll
        for (int r = 0; r < 8; ++r) ms[(h * 8 + r) * 16 + n] = (_Float16)ck[r];
        wave_lds_fence();
        v8h vk = *(const v8h*)&ms[row * 16 + ch * 8];
        *(v8h*)(kh + (size_t)b * L * CQK + (size_t)(l0 + row) * CQK + w * 16 + ch * 8) = vk;
    }
}

// ---------------------------------------------------------------- kernel 2
// Flash attention: 4 waves/block = 4 adjacent 16-query tiles of the same batch.
// 32-key K/V chunks are double-buffered in LDS via async global->LDS copies
// (one staging per block instead of one per wave: 4x less L2 traffic).
__global__ __launch_bounds__(128) void k_attn(const _Float16* __restrict__ qh,
                                              const _Float16* __restrict__ kh,
                                              const _Float16* __restrict__ vTh,
                                              _Float16* __restrict__ oh) {
    __shared__ __align__(16) _Float16 Ks[2][32 * CQK];  // [key][c]  6KB per buf
    __shared__ __align__(16) _Float16 Vs[2][CQK * 32];  // [c][key]  6KB per buf
    __shared__ __align__(16) _Float16 ps[4 * 512];      // per-wave P stage
    const int tid  = threadIdx.x;
    const int lane = tid & 31;
    const int w    = tid >> 5;
    const int wid  = blockIdx.x * 4 + w;   // 4 tiles/block share one batch (256%4==0)
    const int b    = wid >> 8;
    const int l0   = (wid & 255) * 16;
    _Float16* myps = &ps[w * 512];
    const int h = lane >> 4, nl = lane & 15;

    const _Float16* qb = qh + (size_t)b * L * CQK + (size_t)l0 * CQK;
    v16h qf0 = load_a16x32(qb,      CQK);
    v16h qf1 = load_a16x32(qb + 32, CQK);
    v16h qf2 = load_a16x32(qb + 64, CQK);

    v8f acc[6];
#pragma unroll
    for (int t = 0; t < 6; ++t) { v8f z = {}; acc[t] = z; }
    float mrun[8], lrun[8];
#pragma unroll
    for (int r = 0; r < 8; ++r) { mrun[r] = -1e30f; lrun[r] = 0.0f; }

    const char* kgb = (const char*)(kh  + (size_t)b * L * CQK);
    const char* vgb = (const char*)(vTh + (size_t)b * CQK * L);

    // stage one 32-key chunk (K: contiguous 6KB, V: 96 rows x 64B) into buffer
    auto stage = [&](int buf, int lk0) {
        unsigned int kd = (unsigned int)(size_t)(&Ks[buf][0]);
        unsigned int vd = (unsigned int)(size_t)(&Vs[buf][0]);
        const char* ksrc = kgb + (size_t)lk0 * CQK * 2;
#pragma unroll
        for (int i = 0; i < 3; ++i) {               // 384 chunks of 16B
            int idx = i * 128 + tid;
            async_copy16(kd + idx * 16, ksrc + (size_t)idx * 16);
        }
#pragma unroll
        for (int i = 0; i < 3; ++i) {               // 96 rows x 4 chunks
            int idx = i * 128 + tid;
            int row = idx >> 2, sub = idx & 3;
            async_copy16(vd + idx * 16,
                         vgb + ((size_t)row * L + lk0) * 2 + sub * 16);
        }
    };

    stage(0, 0);
    for (int kt = 0; kt < 128; ++kt) {
        const int cur = kt & 1;
        wait_async0();          // own chunk-kt copies done
        __syncthreads();        // everyone's copies done; compute(kt-1) done
        if (kt + 1 < 128) stage(1 - cur, (kt + 1) * 32);

        const _Float16* kk = &Ks[cur][0];
        v8f s0 = {}, s1 = {};
        s0 = wmma_f16(qf0, load_b32x16(kk,            CQK), s0);
        s1 = wmma_f16(qf0, load_b32x16(kk + 16 * CQK, CQK), s1);
        s0 = wmma_f16(qf1, load_b32x16(kk + 32,            CQK), s0);
        s1 = wmma_f16(qf1, load_b32x16(kk + 16 * CQK + 32, CQK), s1);
        s0 = wmma_f16(qf2, load_b32x16(kk + 64,            CQK), s0);
        s1 = wmma_f16(qf2, load_b32x16(kk + 16 * CQK + 64, CQK), s1);

        // streaming softmax: rows live across the 16 lanes of each half
#pragma unroll
        for (int r = 0; r < 8; ++r) {
            float t = fmaxf(s0[r], s1[r]);
            t = fmaxf(t, __shfl_xor(t, 1, 32));
            t = fmaxf(t, __shfl_xor(t, 2, 32));
            t = fmaxf(t, __shfl_xor(t, 4, 32));
            t = fmaxf(t, __shfl_xor(t, 8, 32));
            float mn = fmaxf(mrun[r], t);
            float sc = __expf(mrun[r] - mn);
            mrun[r] = mn;
            lrun[r] *= sc;
#pragma unroll
            for (int tt = 0; tt < 6; ++tt) acc[tt][r] *= sc;
            float p0 = __expf(s0[r] - mn);
            float p1 = __expf(s1[r] - mn);
            float rs = p0 + p1;
            rs += __shfl_xor(rs, 1, 32);
            rs += __shfl_xor(rs, 2, 32);
            rs += __shfl_xor(rs, 4, 32);
            rs += __shfl_xor(rs, 8, 32);
            lrun[r] += rs;
            const int m = h * 8 + r;   // C-layout row
            myps[m * 32 + nl]      = (_Float16)p0;
            myps[m * 32 + 16 + nl] = (_Float16)p1;
        }
        wave_lds_fence();
        // reload P as a 16x32 A-fragment
        v16h pa;
        {
            const _Float16* p = &myps[(lane & 15) * 32 + h * 8];
            v8h lo = *(const v8h*)(p);
            v8h hi = *(const v8h*)(p + 16);
#pragma unroll
            for (int e = 0; e < 8; ++e) { pa[e] = lo[e]; pa[e + 8] = hi[e]; }
        }
#pragma unroll
        for (int t = 0; t < 6; ++t) {
            v16h vf = load_b32x16(&Vs[cur][t * 16 * 32], 32);
            acc[t] = wmma_f16(pa, vf, acc[t]);
        }
        wave_lds_fence();
    }

    // normalize and store out f16 [L][96] coalesced via LDS stage
    float inv[8];
#pragma unroll
    for (int r = 0; r < 8; ++r) inv[r] = 1.0f / lrun[r];
    _Float16* ob = oh + (size_t)b * L * CQK + (size_t)l0 * CQK;
    const int row = lane >> 1, ch = lane & 1;
#pragma unroll
    for (int t = 0; t < 6; ++t) {
#pragma unroll
        for (int r = 0; r < 8; ++r)
            myps[(h * 8 + r) * 16 + nl] = (_Float16)(acc[t][r] * inv[r]);
        wave_lds_fence();
        v8h v = *(const v8h*)&myps[row * 16 + ch * 8];
        *(v8h*)(ob + (size_t)row * CQK + t * 16 + ch * 8) = v;
        wave_lds_fence();
    }
}

// ---------------------------------------------------------------- kernel 3
// y[b,o,l] = gamma * (attn_out @ Wlast^T)[l,o] + x[b,o,l]
__global__ __launch_bounds__(128) void k_final(const _Float16* __restrict__ oh,
                                               const _Float16* __restrict__ wlh,
                                               const float* __restrict__ x,
                                               const float* __restrict__ gamma,
                                               float* __restrict__ y) {
    const int lane = threadIdx.x & 31;
    const int wid  = blockIdx.x * 4 + (threadIdx.x >> 5);
    const int b    = wid / (256 * 12);
    const int rem  = wid % (256 * 12);
    const int l0   = (rem / 12) * 16;
    const int o0   = (rem % 12) * 64;
    const float g  = gamma[0];
    const int h = lane >> 4, nl = lane & 15;

    const _Float16* ab = oh + (size_t)b * L * CQK + (size_t)l0 * CQK;
    v16h a0 = load_a16x32(ab,      CQK);
    v16h a1 = load_a16x32(ab + 32, CQK);
    v16h a2 = load_a16x32(ab + 64, CQK);

#pragma unroll
    for (int nt = 0; nt < 4; ++nt) {
        v8f c = {};
        const _Float16* wb = wlh + (size_t)(o0 + nt * 16) * CQK;
        c = wmma_f16(a0, load_b32x16(wb,      CQK), c);
        c = wmma_f16(a1, load_b32x16(wb + 32, CQK), c);
        c = wmma_f16(a2, load_b32x16(wb + 64, CQK), c);
        const int o = o0 + nt * 16 + nl;
        const size_t off = ((size_t)b * COUT + o) * L + l0 + h * 8;
        const float4* xp = (const float4*)(x + off);
        float4 x0 = xp[0], x1 = xp[1];
        float4 y0, y1;
        y0.x = g * c[0] + x0.x;  y0.y = g * c[1] + x0.y;
        y0.z = g * c[2] + x0.z;  y0.w = g * c[3] + x0.w;
        y1.x = g * c[4] + x1.x;  y1.y = g * c[5] + x1.y;
        y1.z = g * c[6] + x1.z;  y1.w = g * c[7] + x1.w;
        float4* yp = (float4*)(y + off);
        yp[0] = y0;  yp[1] = y1;
    }
}

// ---------------------------------------------------------------- launch
extern "C" void kernel_launch(void* const* d_in, const int* in_sizes, int n_in,
                              void* d_out, int out_size, void* d_ws, size_t ws_size,
                              hipStream_t stream) {
    const float* x  = (const float*)d_in[0];
    const float* Wq = (const float*)d_in[1];
    const float* Wk = (const float*)d_in[2];
    const float* Wv = (const float*)d_in[3];
    const float* Wl = (const float*)d_in[4];
    const float* gm = (const float*)d_in[5];
    float* y = (float*)d_out;

    char* ws = (char*)d_ws;
    const size_t NW  = (size_t)CQK * CIN;        // 73728 per weight matrix
    const size_t NQK = (size_t)NB * L * CQK;     // 3145728
    _Float16* wh  = (_Float16*)ws;               // 4*NW f16
    _Float16* qh  = (_Float16*)(ws + 4 * NW * sizeof(_Float16));
    _Float16* kh  = qh  + NQK;
    _Float16* vTh = kh  + NQK;
    _Float16* oh  = vTh + NQK;

    k_cvt_weights<<<(int)((NW + 255) / 256), 256, 0, stream>>>(Wq, Wk, Wv, Wl, wh);
    k_qkv<<<NB * (L / 16), 192, 0, stream>>>(x, wh, qh, kh, vTh);
    k_attn<<<NB * (L / 16) / 4, 128, 0, stream>>>(qh, kh, vTh, oh);
    k_final<<<NB * (L / 16) * 12 / 4, 128, 0, stream>>>(oh, wh + 3 * NW, x, gm, y);
}
